// TranformerPreprocessed_17076789969482
// MI455X (gfx1250) — compile-verified, matrix operand order checked by reading, not run on hardware
//
#include <hip/hip_runtime.h>
#include <hip/hip_bf16.h>

typedef _Float16 f16_t;
typedef __attribute__((ext_vector_type(16))) _Float16 v16h;
typedef __attribute__((ext_vector_type(8)))  float    v8f;

#define NPTS 32768
#define HH 8
#define DD 24
#define CCN 3
#define LLN 3
#define BBLK 64
#define NBLK (NPTS / BBLK)
#define HD 192
#define MHD 256
#define CHUNK 2048          // LDS-resident sort chunk (16 KB keys+idx)
#define NSORT (2 * LLN * HH) // 48 independent arrays

// ---------------------------------------------------------------------------
// Generic f32 -> f16 convert with zero padding:  src (R,Cs) -> dst (Rp,Cp)
// ---------------------------------------------------------------------------
__global__ void k_pad16(const float* __restrict__ src, f16_t* __restrict__ dst,
                        int R, int Cs, int Rp, int Cp) {
  long i = (long)blockIdx.x * blockDim.x + threadIdx.x;
  long total = (long)Rp * Cp;
  if (i >= total) return;
  int r = (int)(i / Cp), c = (int)(i % Cp);
  float v = (r < R && c < Cs) ? src[(size_t)r * Cs + c] : 0.f;
  dst[i] = (f16_t)v;
}

__global__ void k_zero(float* __restrict__ p, long n) {
  long i = (long)blockIdx.x * blockDim.x + threadIdx.x;
  if (i < n) p[i] = 0.f;
}

// ---------------------------------------------------------------------------
// WMMA GEMM:  C(M,Nc) = act( A(M,K)f16 @ W(Ncp,K)f16^T + bias ) + resid
// blockDim = (32,4): one wave per 16x16 output tile. K multiple of 32.
// ---------------------------------------------------------------------------
__global__ void k_gemm(const f16_t* __restrict__ A, int lda,
                       const f16_t* __restrict__ W, int ldw,
                       const float* __restrict__ bias,
                       const float* __restrict__ resid,
                       float* __restrict__ C,
                       int M, int Nc, int tiles_n, int K, int act) {
  int id = blockIdx.x * 4 + threadIdx.y;          // wave-uniform
  int total = (M >> 4) * tiles_n;
  if (id >= total) return;                        // whole-wave exit: EXEC stays full
  int tn = id % tiles_n;
  int tm = id / tiles_n;
  int lane = threadIdx.x;
  int r15 = lane & 15;
  int koff = (lane >> 4) << 4;
  const f16_t* arow = A + (size_t)(tm * 16 + r15) * lda + koff;
  const f16_t* wrow = W + (size_t)(tn * 16 + r15) * ldw + koff;
  v8f acc = {};
  for (int k = 0; k < K; k += 32) {
    v16h a = *(const v16h*)(arow + k);
    v16h b = *(const v16h*)(wrow + k);
    acc = __builtin_amdgcn_wmma_f32_16x16x32_f16(false, a, false, b, (short)0, acc, false, false);
  }
  int col = tn * 16 + r15;
  int rowadd = (lane >> 4) << 3;
  if (col < Nc) {
    float bv = bias ? bias[col] : 0.f;
    for (int r = 0; r < 8; ++r) {
      int row = tm * 16 + r + rowadd;
      float v = acc[r] + bv;
      if (act == 1) v = fmaxf(v, 0.f);
      size_t o = (size_t)row * Nc + col;
      if (resid) v += resid[o];
      C[o] = v;
    }
  }
}

// ---------------------------------------------------------------------------
// LayerNorm over D=24 (one thread per row)
// ---------------------------------------------------------------------------
__global__ void k_ln24(const float* __restrict__ x, const float* __restrict__ g,
                       const float* __restrict__ b, float* __restrict__ y, int Nn) {
  int n = blockIdx.x * blockDim.x + threadIdx.x;
  if (n >= Nn) return;
  const float* r = x + (size_t)n * DD;
  float* o = y + (size_t)n * DD;
  float m = 0.f;
  for (int d = 0; d < DD; ++d) m += r[d];
  m *= (1.f / DD);
  float v = 0.f;
  for (int d = 0; d < DD; ++d) { float t = r[d] - m; v += t * t; }
  v *= (1.f / DD);
  float inv = rsqrtf(v + 1e-5f);
  for (int d = 0; d < DD; ++d) o[d] = (r[d] - m) * inv * g[d] + b[d];
}

// ---------------------------------------------------------------------------
// RPE scale:  sw[h,c] = sqrt(2 * qw[h,c]),  qw cols map c->{0,0,1}
// ---------------------------------------------------------------------------
__global__ void k_sw(const float* __restrict__ w_rpe, float* __restrict__ sw) {
  __shared__ float tmp[16];
  int t = threadIdx.x;
  if (t < 16) {
    int h = t >> 1, cc = t & 1;
    float acc = 0.f;
    for (int nw = 0; nw < 4; ++nw) {
      float s = 0.f;
      for (int d = 0; d < DD; ++d) s += w_rpe[((size_t)(h * DD + d)) * 8 + cc * 4 + nw];
      acc += __expf(fminf(s, 50.f));
    }
    tmp[t] = acc;
  }
  __syncthreads();
  if (t < HH * CCN) {
    int h = t / CCN, c = t % CCN;
    int cc = (c == 0) ? 0 : (c - 1);
    sw[t] = sqrtf(2.f * tmp[h * 2 + cc]);
  }
}

// ---------------------------------------------------------------------------
// Build q_hat/k_hat (H,N,32) f32+f16 (cols 0..23 proj, 24..26 swr, 27..31 zero)
// plus -0.5*||.||^2 over the 27 real dims.
// ---------------------------------------------------------------------------
__global__ void k_build_hat(const float* __restrict__ src192, const float* __restrict__ coords,
                            const float* __restrict__ sw, float* __restrict__ hatf,
                            f16_t* __restrict__ hath, float* __restrict__ sqsum, int Nn) {
  int idx = blockIdx.x * blockDim.x + threadIdx.x;
  if (idx >= Nn * HH) return;
  int h = idx / Nn, n = idx % Nn;
  float s = 0.f;
  float* of = hatf + ((size_t)h * Nn + n) * 32;
  f16_t* oh = hath + ((size_t)h * Nn + n) * 32;
  for (int d = 0; d < DD; ++d) {
    float v = src192[(size_t)n * HD + h * DD + d];
    of[d] = v; oh[d] = (f16_t)v; s += v * v;
  }
  for (int c = 0; c < CCN; ++c) {
    float v = sw[h * CCN + c] * coords[(size_t)n * CCN + c];
    of[DD + c] = v; oh[DD + c] = (f16_t)v; s += v * v;
  }
  for (int d = DD + CCN; d < 32; ++d) { of[d] = 0.f; oh[d] = (f16_t)0.f; }
  sqsum[(size_t)h * Nn + n] = -0.5f * s;
}

__global__ void k_build_vhat(const float* __restrict__ src192, f16_t* __restrict__ hath, int Nn) {
  int idx = blockIdx.x * blockDim.x + threadIdx.x;
  if (idx >= Nn * HH) return;
  int h = idx / Nn, n = idx % Nn;
  f16_t* oh = hath + ((size_t)h * Nn + n) * 32;
  for (int d = 0; d < DD; ++d) oh[d] = (f16_t)src192[(size_t)n * HD + h * DD + d];
  for (int d = DD; d < 32; ++d) oh[d] = (f16_t)0.f;
}

// ---------------------------------------------------------------------------
// LSH hash: out[l,h,n] = sum_d hat[h,n,d] * alpha[h,d,l]   (d over 27)
// ---------------------------------------------------------------------------
__global__ void k_hash(const float* __restrict__ hatf, const float* __restrict__ alpha,
                       float* __restrict__ out, int Nn) {
  int idx = blockIdx.x * blockDim.x + threadIdx.x;
  if (idx >= LLN * HH * Nn) return;
  int n = idx % Nn, t = idx / Nn;
  int h = t % HH, l = t / HH;
  const float* row = hatf + ((size_t)h * Nn + n) * 32;
  float s = 0.f;
  for (int d = 0; d < DD + CCN; ++d) s += row[d] * alpha[(size_t)(h * (DD + CCN) + d) * LLN + l];
  out[((size_t)l * HH + h) * Nn + n] = s;
}

// ---------------------------------------------------------------------------
// hash_shift[l,h] = max(max q, max k) - min(min q, min k) over n
// ---------------------------------------------------------------------------
__global__ void k_shift(const float* __restrict__ qh, const float* __restrict__ kh,
                        float* __restrict__ shift, int Nn) {
  __shared__ float smax[256], smin[256];
  int lh = blockIdx.x, tid = threadIdx.x;
  const float* a = qh + (size_t)lh * Nn;
  const float* b = kh + (size_t)lh * Nn;
  float mx = -3.4e38f, mn = 3.4e38f;
  for (int i = tid; i < Nn; i += 256) {
    float x = a[i], y = b[i];
    mx = fmaxf(mx, fmaxf(x, y));
    mn = fminf(mn, fminf(x, y));
  }
  smax[tid] = mx; smin[tid] = mn; __syncthreads();
  for (int s = 128; s > 0; s >>= 1) {
    if (tid < s) { smax[tid] = fmaxf(smax[tid], smax[tid + s]); smin[tid] = fminf(smin[tid], smin[tid + s]); }
    __syncthreads();
  }
  if (tid == 0) shift[lh] = smax[0] - smin[0];
}

__global__ void k_cs_init(const int* __restrict__ cshift, float* __restrict__ cs, int Nn) {
  int idx = blockIdx.x * blockDim.x + threadIdx.x;
  if (idx >= LLN * HH * Nn) return;
  cs[idx] = (float)cshift[idx % Nn];
}

// cs *= shift ; build sort keys (q side then k side) and init indices
__global__ void k_cs_keys(float* __restrict__ cs, const float* __restrict__ shift,
                          const float* __restrict__ qh, const float* __restrict__ kh,
                          float* __restrict__ keys, int* __restrict__ idxs, int Nn) {
  int idx = blockIdx.x * blockDim.x + threadIdx.x;
  if (idx >= LLN * HH * Nn) return;
  int n = idx % Nn, lh = idx / Nn;
  size_t o = (size_t)lh * Nn + n;
  size_t half = (size_t)LLN * HH * Nn;
  float c = cs[o] * shift[lh];
  cs[o] = c;
  keys[o] = qh[o] + c;
  keys[half + o] = kh[o] + c;
  idxs[o] = n;
  idxs[half + o] = n;
}

// ---------------------------------------------------------------------------
// Bitonic sort: LDS-resident local phases + sparse global cross-chunk steps.
// ---------------------------------------------------------------------------
__device__ __forceinline__ void bitonic_lds_pass(float* kk_, int* ii_, int base,
                                                 int kk2, int jstart) {
  int t = threadIdx.x;
  for (int j = jstart; j > 0; j >>= 1) {
    for (int li = t; li < CHUNK; li += 1024) {
      int ixj = li ^ j;
      if (ixj > li) {
        bool up = (((base + li) & kk2) == 0);
        float a = kk_[li], c = kk_[ixj];
        if (up ? (a > c) : (a < c)) {
          kk_[li] = c; kk_[ixj] = a;
          int tt = ii_[li]; ii_[li] = ii_[ixj]; ii_[ixj] = tt;
        }
      }
    }
    __syncthreads();
  }
}

// Full local sort of aligned 2048-chunks (all kk<=CHUNK stages)
__global__ void k_bitonic_local_sort(float* __restrict__ keys, int* __restrict__ idxs, int Nn) {
  __shared__ float kk_[CHUNK];
  __shared__ int ii_[CHUNK];
  int b = blockIdx.y;
  int base = blockIdx.x * CHUNK;
  float* K = keys + (size_t)b * Nn + base;
  int* I = idxs + (size_t)b * Nn + base;
  int t = threadIdx.x;
  kk_[t] = K[t]; kk_[t + 1024] = K[t + 1024];
  ii_[t] = I[t]; ii_[t + 1024] = I[t + 1024];
  __syncthreads();
  for (int kk2 = 2; kk2 <= CHUNK; kk2 <<= 1)
    bitonic_lds_pass(kk_, ii_, base, kk2, kk2 >> 1);
  K[t] = kk_[t]; K[t + 1024] = kk_[t + 1024];
  I[t] = ii_[t]; I[t + 1024] = ii_[t + 1024];
}

// Local merge tail: stages j = CHUNK/2 .. 1 of outer stage kk2 (>CHUNK)
__global__ void k_bitonic_local_merge(float* __restrict__ keys, int* __restrict__ idxs,
                                      int kk2, int Nn) {
  __shared__ float kk_[CHUNK];
  __shared__ int ii_[CHUNK];
  int b = blockIdx.y;
  int base = blockIdx.x * CHUNK;
  float* K = keys + (size_t)b * Nn + base;
  int* I = idxs + (size_t)b * Nn + base;
  int t = threadIdx.x;
  kk_[t] = K[t]; kk_[t + 1024] = K[t + 1024];
  ii_[t] = I[t]; ii_[t + 1024] = I[t + 1024];
  __syncthreads();
  bitonic_lds_pass(kk_, ii_, base, kk2, CHUNK >> 1);
  K[t] = kk_[t]; K[t + 1024] = kk_[t + 1024];
  I[t] = ii_[t]; I[t + 1024] = ii_[t + 1024];
}

// Global cross-chunk compare-exchange step (j >= CHUNK)
__global__ void k_bitonic(float* __restrict__ keys, int* __restrict__ idxs,
                          int kk, int j, int Nn) {
  int i = blockIdx.x * blockDim.x + threadIdx.x;
  int b = blockIdx.y;
  int ixj = i ^ j;
  if (ixj <= i) return;
  float* K = keys + (size_t)b * Nn;
  int* I = idxs + (size_t)b * Nn;
  float a = K[i], c = K[ixj];
  bool up = ((i & kk) == 0);
  if (up ? (a > c) : (a < c)) {
    K[i] = c; K[ixj] = a;
    int t = I[i]; I[i] = I[ixj]; I[ixj] = t;
  }
}

// ---------------------------------------------------------------------------
// Block attention (B=64), one workgroup per (block, h, l). 8 waves.
// Gather q/k tiles via GLOBAL_LOAD_ASYNC_TO_LDS_B128 (ASYNCcnt), v via
// transposing scalar copies. S = sq@sk^T (WMMA), P = exp(min(S+qsq+ksq,0)),
// O = P@sv (WMMA), scatter O / denominators back through qpos with atomics.
// ---------------------------------------------------------------------------
__global__ void k_attn(const f16_t* __restrict__ qhath, const f16_t* __restrict__ khath,
                       const f16_t* __restrict__ vhath, const float* __restrict__ qsqb,
                       const float* __restrict__ ksqb, const int* __restrict__ qpos,
                       const int* __restrict__ kpos, float* __restrict__ oacc,
                       float* __restrict__ dacc, int Nn) {
  __shared__ __align__(32) f16_t sq[BBLK * 32];
  __shared__ __align__(32) f16_t sk[BBLK * 32];
  __shared__ __align__(32) f16_t svT[32 * BBLK];
  __shared__ __align__(32) f16_t pm[BBLK * BBLK];
  __shared__ float qs[BBLK], ksum[BBLK];
  int nb = blockIdx.x, h = blockIdx.y, l = blockIdx.z;
  int tid = threadIdx.x;
  const int* qp = qpos + ((size_t)(l * HH + h)) * Nn + nb * BBLK;
  const int* kp = kpos + ((size_t)(l * HH + h)) * Nn + nb * BBLK;
  int row = tid >> 2, q4 = (tid & 3) * 8;
  int qi = qp[row], ki = kp[row];
  const f16_t* qsrc = qhath + ((size_t)h * Nn + qi) * 32 + q4;
  const f16_t* ksrc = khath + ((size_t)h * Nn + ki) * 32 + q4;
  const f16_t* vsrc = vhath + ((size_t)h * Nn + ki) * 32 + q4;
  {
    // async gather: 16B per lane, per-lane LDS byte address (generic ptr low bits)
    unsigned lq = (unsigned)(size_t)(&sq[row * 32 + q4]);
    unsigned lk = (unsigned)(size_t)(&sk[row * 32 + q4]);
    unsigned long long gq = (unsigned long long)(size_t)qsrc;
    unsigned long long gk = (unsigned long long)(size_t)ksrc;
    asm volatile("global_load_async_to_lds_b128 %0, %1, off" :: "v"(lq), "v"(gq) : "memory");
    asm volatile("global_load_async_to_lds_b128 %0, %1, off" :: "v"(lk), "v"(gk) : "memory");
  }
  for (int e = 0; e < 8; ++e) svT[(q4 + e) * BBLK + row] = vsrc[e];
  if (tid < BBLK) {
    qs[tid] = qsqb[(size_t)h * Nn + qp[tid]];
    ksum[tid] = ksqb[(size_t)h * Nn + kp[tid]];
  }
  asm volatile("s_wait_asynccnt 0" ::: "memory");
  __syncthreads();
  int wave = tid >> 5, lane = tid & 31, r15 = lane & 15;
  int koff = (lane >> 4) << 4, rowadd = (lane >> 4) << 3;
  // ---- S = sq @ sk^T : 16 tiles, 2 per wave ----
  for (int t = wave; t < 16; t += 8) {
    int tm = t >> 2, tn = t & 3;
    v16h a = *(const v16h*)(&sq[(tm * 16 + r15) * 32 + koff]);
    v16h b = *(const v16h*)(&sk[(tn * 16 + r15) * 32 + koff]);
    v8f acc = {};
    acc = __builtin_amdgcn_wmma_f32_16x16x32_f16(false, a, false, b, (short)0, acc, false, false);
    int cc = tn * 16 + r15;
    for (int r = 0; r < 8; ++r) {
      int rr = tm * 16 + r + rowadd;
      float v = acc[r] + qs[rr] + ksum[cc];
      pm[rr * BBLK + cc] = (f16_t)__expf(fminf(v, 0.f));
    }
  }
  __syncthreads();
  // ---- per-row denominators -> dacc (scatter by qpos) ----
  if (tid < BBLK) {
    float s = 1e-20f;
    for (int j2 = 0; j2 < BBLK; ++j2) s += (float)pm[tid * BBLK + j2];
    atomicAdd(&dacc[(size_t)h * Nn + qp[tid]], s);
  }
  // ---- O = P @ sv : 8 tiles (4x2), one per wave ----
  {
    int tm = wave >> 1, tn = wave & 1;
    v8f acc = {};
    for (int k2 = 0; k2 < BBLK; k2 += 32) {
      v16h a = *(const v16h*)(&pm[(tm * 16 + r15) * BBLK + k2 + koff]);
      v16h b = *(const v16h*)(&svT[(tn * 16 + r15) * BBLK + k2 + koff]);
      acc = __builtin_amdgcn_wmma_f32_16x16x32_f16(false, a, false, b, (short)0, acc, false, false);
    }
    int cc = tn * 16 + r15;
    if (cc < DD) {
      for (int r = 0; r < 8; ++r) {
        int rr = tm * 16 + r + rowadd;
        atomicAdd(&oacc[((size_t)h * Nn + qp[rr]) * DD + cc], acc[r]);
      }
    }
  }
}

// ---------------------------------------------------------------------------
// aggr[n, j] with the reference's (H,N,D).reshape(N, H*D) index mixing
// ---------------------------------------------------------------------------
__global__ void k_combine(const float* __restrict__ oacc, const float* __restrict__ dacc,
                          f16_t* __restrict__ aggr, int Nn) {
  long i = (long)blockIdx.x * blockDim.x + threadIdx.x;
  long total = (long)Nn * HD;
  if (i >= total) return;
  long ND = (long)Nn * DD;
  int hp = (int)(i / ND);
  long rem = i % ND;
  int np_ = (int)(rem / DD), dp = (int)(rem % DD);
  float v = oacc[((size_t)hp * Nn + np_) * DD + dp] / dacc[(size_t)hp * Nn + np_];
  aggr[i] = (f16_t)v;
}

// ---------------------------------------------------------------------------
// Fused LayerNorm(256) + tanh -> f16
// ---------------------------------------------------------------------------
__global__ void k_ln_tanh(const float* __restrict__ x, const float* __restrict__ g,
                          const float* __restrict__ b, f16_t* __restrict__ y, int Nn) {
  __shared__ float red[MHD];
  __shared__ float mv[2];
  int n = blockIdx.x, c = threadIdx.x;
  float v = x[(size_t)n * MHD + c];
  red[c] = v; __syncthreads();
  for (int s = 128; s > 0; s >>= 1) { if (c < s) red[c] += red[c + s]; __syncthreads(); }
  if (c == 0) mv[0] = red[0] * (1.f / MHD);
  __syncthreads();
  float d = v - mv[0];
  red[c] = d * d; __syncthreads();
  for (int s = 128; s > 0; s >>= 1) { if (c < s) red[c] += red[c + s]; __syncthreads(); }
  if (c == 0) mv[1] = red[0] * (1.f / MHD);
  __syncthreads();
  float inv = rsqrtf(mv[1] + 1e-5f);
  y[(size_t)n * MHD + c] = (f16_t)tanhf(d * inv * g[c] + b[c]);
}

__global__ void k_concat(const float* __restrict__ x0, const float* __restrict__ x1,
                         const float* __restrict__ x2, f16_t* __restrict__ dst, int Nn) {
  long i = (long)blockIdx.x * blockDim.x + threadIdx.x;
  long total = (long)Nn * 96;
  if (i >= total) return;
  int n = (int)(i / 96), c = (int)(i % 96);
  float v = 0.f;
  if (c < 24) v = x0[(size_t)n * DD + c];
  else if (c < 48) v = x1[(size_t)n * DD + c - 24];
  else if (c < 72) v = x2[(size_t)n * DD + c - 48];
  dst[i] = (f16_t)v;
}

// ===========================================================================
// Host side
// ===========================================================================
struct Work {
  float *x0, *x1, *x2, *cs, *t0, *t1, *qb, *kb, *vb, *sw, *shift;
  float *qhatf, *khatf, *qsq, *ksq, *qh, *kh, *keys, *oacc, *dacc;
  int* idxs;
  f16_t *qhath, *khath, *vhath, *af, *wf;
};

static void conv_pad(const float* src, f16_t* dst, int R, int Cs, int Rp, int Cp, hipStream_t s) {
  long total = (long)Rp * Cp;
  k_pad16<<<dim3((unsigned)((total + 255) / 256)), 256, 0, s>>>(src, dst, R, Cs, Rp, Cp);
}

static void gemm(const f16_t* A, int lda, const f16_t* W, int ldw, const float* bias,
                 const float* resid, float* C, int M, int Nc, int Ncp, int K, int act,
                 hipStream_t s) {
  int tiles_n = Ncp / 16;
  int total = (M / 16) * tiles_n;
  k_gemm<<<dim3((unsigned)((total + 3) / 4)), dim3(32, 4), 0, s>>>(
      A, lda, W, ldw, bias, resid, C, M, Nc, tiles_n, K, act);
}

// P[0..14] = ln1_g, ln1_b, wq, wk, wv, w_rpe, alpha, out_w, out_b,
//            ln2_g, ln2_b, ff_w1, ff_b1, ff_w2, ff_b2
static void run_layer(const Work& w, const float* const P[15], const float* coords,
                      const float* xin, float* xout, hipStream_t s) {
  const int N = NPTS;
  const size_t half = (size_t)LLN * HH * N;
  // LN1 + QKV projections (WMMA)
  k_ln24<<<(N + 255) / 256, 256, 0, s>>>(xin, P[0], P[1], w.t0, N);
  conv_pad(w.t0, w.af, N, DD, N, 32, s);
  conv_pad(P[2], w.wf, HD, DD, HD, 32, s);
  gemm(w.af, 32, w.wf, 32, nullptr, nullptr, w.qb, N, HD, HD, 32, 0, s);
  conv_pad(P[3], w.wf, HD, DD, HD, 32, s);
  gemm(w.af, 32, w.wf, 32, nullptr, nullptr, w.kb, N, HD, HD, 32, 0, s);
  conv_pad(P[4], w.wf, HD, DD, HD, 32, s);
  gemm(w.af, 32, w.wf, 32, nullptr, nullptr, w.vb, N, HD, HD, 32, 0, s);
  // RPE scales + hat construction
  k_sw<<<1, 32, 0, s>>>(P[5], w.sw);
  k_build_hat<<<(N * HH + 255) / 256, 256, 0, s>>>(w.qb, coords, w.sw, w.qhatf, w.qhath, w.qsq, N);
  k_build_hat<<<(N * HH + 255) / 256, 256, 0, s>>>(w.kb, coords, w.sw, w.khatf, w.khath, w.ksq, N);
  k_build_vhat<<<(N * HH + 255) / 256, 256, 0, s>>>(w.vb, w.vhath, N);
  // LSH hashing + shift + keys
  k_hash<<<(LLN * HH * N + 255) / 256, 256, 0, s>>>(w.qhatf, P[6], w.qh, N);
  k_hash<<<(LLN * HH * N + 255) / 256, 256, 0, s>>>(w.khatf, P[6], w.kh, N);
  k_shift<<<LLN * HH, 256, 0, s>>>(w.qh, w.kh, w.shift, N);
  k_cs_keys<<<(LLN * HH * N + 255) / 256, 256, 0, s>>>(w.cs, w.shift, w.qh, w.kh, w.keys, w.idxs, N);
  // batched bitonic argsort of 2*L*H arrays: LDS local sort + sparse global steps
  k_bitonic_local_sort<<<dim3(N / CHUNK, NSORT), 1024, 0, s>>>(w.keys, w.idxs, N);
  for (int kk2 = CHUNK * 2; kk2 <= N; kk2 <<= 1) {
    for (int j = kk2 >> 1; j >= CHUNK; j >>= 1)
      k_bitonic<<<dim3(N / 256, NSORT), 256, 0, s>>>(w.keys, w.idxs, kk2, j, N);
    k_bitonic_local_merge<<<dim3(N / CHUNK, NSORT), 1024, 0, s>>>(w.keys, w.idxs, kk2, N);
  }
  // block attention, accumulated over L via atomics
  k_zero<<<(unsigned)(((long)HH * N * DD + 255) / 256), 256, 0, s>>>(w.oacc, (long)HH * N * DD);
  k_zero<<<(unsigned)(((long)HH * N + 255) / 256), 256, 0, s>>>(w.dacc, (long)HH * N);
  k_attn<<<dim3(NBLK, HH, LLN), 256, 0, s>>>(w.qhath, w.khath, w.vhath, w.qsq, w.ksq,
                                             w.idxs, w.idxs + half, w.oacc, w.dacc, N);
  k_combine<<<(unsigned)(((long)N * HD + 255) / 256), 256, 0, s>>>(w.oacc, w.dacc, w.af, N);
  // out projection with residual
  conv_pad(P[7], w.wf, DD, HD, 32, HD, s);
  gemm(w.af, HD, w.wf, HD, P[8], xin, xout, N, DD, 32, HD, 0, s);
  // FFN with residual
  k_ln24<<<(N + 255) / 256, 256, 0, s>>>(xout, P[9], P[10], w.t0, N);
  conv_pad(w.t0, w.af, N, DD, N, 32, s);
  conv_pad(P[11], w.wf, DD, DD, 32, 32, s);
  gemm(w.af, 32, w.wf, 32, P[12], nullptr, w.t1, N, DD, 32, 32, 1, s);
  conv_pad(w.t1, w.af, N, DD, N, 32, s);
  conv_pad(P[13], w.wf, DD, DD, 32, 32, s);
  gemm(w.af, 32, w.wf, 32, P[14], xout, xout, N, DD, 32, 32, 0, s);
}

extern "C" void kernel_launch(void* const* d_in, const int* in_sizes, int n_in,
                              void* d_out, int out_size, void* d_ws, size_t ws_size,
                              hipStream_t stream) {
  (void)in_sizes; (void)n_in; (void)out_size; (void)ws_size;
  const int N = NPTS;
  auto F = [&](int i) -> const float* { return (const float*)d_in[i]; };
  const float* x_in   = F(0);
  const int*   cshift = (const int*)d_in[1];
  const float* coords = F(2);
  // d_in[3] = unpad_seq (unused by reference)
  const float* fe_w1 = F(4); const float* fe_b1 = F(5);
  const float* fe_w2 = F(6); const float* fe_b2 = F(7);
  const float* P1[15]; const float* P2[15];
  for (int i = 0; i < 15; ++i) { P1[i] = F(8 + i); P2[i] = F(23 + i); }
  const float* W_w = F(38);
  // mo blocks: F(39+4i)=w, +1=b, +2=g, +3=bb ; F(55)=mo_w4, F(56)=mo_b4
  const float* op_w = F(57); const float* op_b = F(58);

  // ---- workspace bump allocator ----
  char* base = (char*)d_ws;
  size_t off = 0;
  auto alloc = [&](size_t bytes) -> void* {
    void* p = base + off;
    off = (off + bytes + 255) & ~(size_t)255;
    return p;
  };
  Work w;
  w.x0 = (float*)alloc((size_t)N * DD * 4);
  w.x1 = (float*)alloc((size_t)N * DD * 4);
  w.x2 = (float*)alloc((size_t)N * DD * 4);
  w.cs = (float*)alloc((size_t)LLN * HH * N * 4);
  w.t0 = (float*)alloc((size_t)N * MHD * 4);
  w.t1 = (float*)alloc((size_t)N * MHD * 4);
  w.qb = (float*)alloc((size_t)N * HD * 4);
  w.kb = (float*)alloc((size_t)N * HD * 4);
  w.vb = (float*)alloc((size_t)N * HD * 4);
  w.sw = (float*)alloc(64 * 4);
  w.shift = (float*)alloc(64 * 4);
  w.qhatf = (float*)alloc((size_t)HH * N * 32 * 4);
  w.khatf = (float*)alloc((size_t)HH * N * 32 * 4);
  w.qsq = (float*)alloc((size_t)HH * N * 4);
  w.ksq = (float*)alloc((size_t)HH * N * 4);
  w.qh = (float*)alloc((size_t)LLN * HH * N * 4);
  w.kh = (float*)alloc((size_t)LLN * HH * N * 4);
  w.keys = (float*)alloc((size_t)2 * LLN * HH * N * 4);
  w.oacc = (float*)alloc((size_t)HH * N * DD * 4);
  w.dacc = (float*)alloc((size_t)HH * N * 4);
  w.idxs = (int*)alloc((size_t)2 * LLN * HH * N * 4);
  w.qhath = (f16_t*)alloc((size_t)HH * N * 32 * 2);
  w.khath = (f16_t*)alloc((size_t)HH * N * 32 * 2);
  w.vhath = (f16_t*)alloc((size_t)HH * N * 32 * 2);
  w.af = (f16_t*)alloc((size_t)N * MHD * 2);
  w.wf = (f16_t*)alloc((size_t)MHD * MHD * 2);

  // ---- cs init (broadcast combined_shifts) ----
  k_cs_init<<<(LLN * HH * N + 255) / 256, 256, 0, stream>>>(cshift, w.cs, N);

  // ---- feature encoder: relu(x@fe_w1^T+b1)@fe_w2^T+b2 ----
  conv_pad(x_in, w.af, N, 32, N, 32, stream);
  conv_pad(fe_w1, w.wf, DD, 32, 32, 32, stream);
  gemm(w.af, 32, w.wf, 32, fe_b1, nullptr, w.t0, N, DD, 32, 32, 1, stream);
  conv_pad(w.t0, w.af, N, DD, N, 32, stream);
  conv_pad(fe_w2, w.wf, DD, DD, 32, 32, stream);
  gemm(w.af, 32, w.wf, 32, fe_b2, nullptr, w.x0, N, DD, 32, 32, 0, stream);

  // ---- two HEPT layers ----
  run_layer(w, P1, coords, w.x0, w.x1, stream);
  run_layer(w, P2, coords, w.x1, w.x2, stream);

  // ---- head: concat -> W_w -> 4x(lin+LN+tanh) -> mo_w4 -> op ----
  k_concat<<<(unsigned)(((long)N * 96 + 255) / 256), 256, 0, stream>>>(w.x0, w.x1, w.x2, w.af, N);
  conv_pad(W_w, w.wf, 12, 72, 16, 96, stream);
  gemm(w.af, 96, w.wf, 96, nullptr, nullptr, w.t0, N, 12, 16, 96, 0, stream);
  conv_pad(w.t0, w.af, N, 12, N, 32, stream);
  int K = 32;
  for (int i = 0; i < 4; ++i) {
    conv_pad(F(39 + 4 * i), w.wf, MHD, (i == 0 ? 12 : MHD), MHD, K, stream);
    gemm(w.af, K, w.wf, K, F(39 + 4 * i + 1), nullptr, w.t1, N, MHD, MHD, K, 0, stream);
    k_ln_tanh<<<N, MHD, 0, stream>>>(w.t1, F(39 + 4 * i + 2), F(39 + 4 * i + 3), w.af, N);
    K = MHD;
  }
  conv_pad(F(55), w.wf, 12, MHD, 16, MHD, stream);
  gemm(w.af, MHD, w.wf, MHD, F(56), nullptr, w.t0, N, 12, 16, MHD, 0, stream);
  conv_pad(w.t0, w.af, N, 12, N, 32, stream);
  conv_pad(op_w, w.wf, 8, 12, 16, 32, stream);
  gemm(w.af, 32, w.wf, 32, op_b, nullptr, (float*)d_out, N, 8, 16, 32, 0, stream);
}
